// Seq2SeqAttn_13374528159868
// MI455X (gfx1250) — matrix operand order repaired and checked by reading, not compiled, not run on hardware
//
#include <hip/hip_runtime.h>

typedef __bf16 bf16;
typedef __attribute__((ext_vector_type(16))) __bf16 v16bf;
typedef __attribute__((ext_vector_type(8)))  float  v8f;

#define WMMA_BF16(a, b, c) \
  __builtin_amdgcn_wmma_f32_16x16x32_bf16(false, (a), false, (b), (short)0, (c), false, false)

// ---- model dims ----
#define NB    32      // batch
#define TENC  128
#define TDEC  32
#define ESRC  256
#define ETGT  256
#define HENC  512
#define G4    2048    // 4*HENC (lstm gates)
#define CTXD  1024    // 2*HENC
#define HDEC  512
#define G3    1536    // 3*HDEC (gru gates)
#define KIN   1280    // ETGT + CTXD
#define VTGT  32000
#define ATT   8

__device__ __forceinline__ float sigf(float x) { return 1.0f / (1.0f + expf(-x)); }

// ------------------------------------------------------------------
// WMMA fragment index maps (wave32 layouts per CDNA5 ISA 7.12.2).
// Packed-A: [Mtile][Ktile][lane 0..31][elem 0..15] bf16 (32 B per lane)
// Packed-B: [Ktile][Ntile][lane 0..31][elem 0..15] bf16
// C/D 16x16 f32: reg r -> M = r + 8*(lane>=16), N = lane&15
// ------------------------------------------------------------------

// (m,k) -> flat bf16 slot in packed-A (Ktiles = K/32)
__device__ __forceinline__ int a_slot(int m, int k, int Ktiles) {
  int mt = m >> 4, kt = k >> 5, kk = k & 31;
  int hi = (kk >> 3) & 1;
  int lane = (m & 15) + (hi << 4);
  int kkp = kk - (hi << 3);
  int v = ((kkp & 16) >> 2) + ((kkp & 7) >> 1);
  int e = (v << 1) + (kk & 1);
  return (((mt * Ktiles + kt) << 5) + lane) * 16 + e;
}

// flat packed-A slot -> (m,k)
__device__ __forceinline__ void a_coord(int i, int Ktiles, int& m, int& k) {
  int e = i & 15, lane = (i >> 4) & 31, frag = i >> 9;
  int mt = frag / Ktiles, kt = frag - mt * Ktiles;
  m = (mt << 4) + (lane & 15);
  int v = e >> 1;
  k = (kt << 5) + ((v & 4) << 2) + ((v & 3) << 1) + (e & 1) + ((lane & 16) ? 8 : 0);
}

// ------------------------------------------------------------------
// Pack W[N,K] f32 row-major -> packed-B fragments (transpose+convert+swizzle)
// ------------------------------------------------------------------
__global__ __launch_bounds__(256) void k_packB(const float* __restrict__ src,
                                               bf16* __restrict__ dst, int N, int K) {
  int i = blockIdx.x * 256 + threadIdx.x;              // over K*N
  if (i >= N * K) return;
  int e = i & 15, lane = (i >> 4) & 31, frag = i >> 9;
  int Ntiles = N >> 4;
  int kt = frag / Ntiles, nt = frag - kt * Ntiles;
  int k = (kt << 5) + ((lane & 16) ? 8 : 0) + ((e & 8) << 1) + (e & 7);
  int n = (nt << 4) + (lane & 15);
  dst[i] = (bf16)src[(size_t)n * K + k];
}

// ------------------------------------------------------------------
// Pack X[M,K] f32 row-major -> packed-A fragments
// ------------------------------------------------------------------
__global__ __launch_bounds__(256) void k_packA(const float* __restrict__ src,
                                               bf16* __restrict__ dst, int M, int K) {
  int i = blockIdx.x * 256 + threadIdx.x;              // over M*K
  if (i >= M * K) return;
  int m, k;
  a_coord(i, K >> 5, m, k);
  dst[i] = (bf16)src[(size_t)m * K + k];
}

// ------------------------------------------------------------------
// Encoder embedding gather written DIRECTLY in packed-A layout
// (rows m = t*NB+b); also per-sequence reversed copy.
// ------------------------------------------------------------------
__global__ __launch_bounds__(256) void k_embed_pack(const int* __restrict__ toks,
                                                    const int* __restrict__ lens,
                                                    const float* __restrict__ emb,
                                                    bf16* __restrict__ xA,
                                                    bf16* __restrict__ xrA) {
  int i = blockIdx.x * 256 + threadIdx.x;              // TENC*NB*ESRC
  int m, col;
  a_coord(i, ESRC >> 5, m, col);
  int t = m >> 5, b = m & (NB - 1);
  int len = lens[b];
  int rt = (t < len) ? (len - 1 - t) : t;
  xA[i]  = (bf16)emb[(size_t)toks[b * TENC + t]  * ESRC + col];
  xrA[i] = (bf16)emb[(size_t)toks[b * TENC + rt] * ESRC + col];
}

// ------------------------------------------------------------------
// Xg[4096,2048] = x @ WihT + (bih+bhh)   4Mx2N register-blocked WMMA
// jobs: 64 Mgroups x 64 Ngroups = 4096 waves (512 blocks)
// ------------------------------------------------------------------
__global__ __launch_bounds__(256) void k_gemm_xg(const bf16* __restrict__ xA,
                                                 const bf16* __restrict__ WP,
                                                 const float* __restrict__ bih,
                                                 const float* __restrict__ bhh,
                                                 float* __restrict__ out) {
  const int wave = threadIdx.x >> 5, L = threadIdx.x & 31;
  const v16bf* Av = (const v16bf*)xA;
  const v16bf* Bv = (const v16bf*)WP;
  int job = blockIdx.x * 8 + wave;
  int mg = job & 63, ng = job >> 6;                    // 4 Mtiles, 2 Ntiles
  int mtb = mg << 2, ntb = ng << 1;
  v8f c[4][2] = {};
#pragma unroll 2
  for (int kt = 0; kt < ESRC / 32; ++kt) {
    v16bf b0 = Bv[((kt * (G4 / 16) + ntb) << 5) + L];
    v16bf b1 = Bv[((kt * (G4 / 16) + ntb + 1) << 5) + L];
#pragma unroll
    for (int im = 0; im < 4; ++im) {
      v16bf a = Av[(((mtb + im) * (ESRC / 32) + kt) << 5) + L];
      c[im][0] = WMMA_BF16(a, b0, c[im][0]);
      c[im][1] = WMMA_BF16(a, b1, c[im][1]);
    }
  }
#pragma unroll
  for (int jn = 0; jn < 2; ++jn) {
    int n = ((ntb + jn) << 4) + (L & 15);
    float bias = bih[n] + bhh[n];
#pragma unroll
    for (int im = 0; im < 4; ++im)
#pragma unroll
      for (int r = 0; r < 8; ++r) {
        int m = ((mtb + im) << 4) + r + ((L & 16) ? 8 : 0);
        out[(size_t)m * G4 + n] = c[im][jn][r] + bias;
      }
  }
}

// ------------------------------------------------------------------
// Persistent LSTM recurrence, one block per direction.
// h in LDS packed-A bf16; c in registers.  2Mx2N blocking per wave-job.
// ------------------------------------------------------------------
__global__ __launch_bounds__(512) void k_lstm(const float* __restrict__ Xg_f,
                                              const float* __restrict__ Xg_b,
                                              const bf16* __restrict__ WhhP_f,
                                              const bf16* __restrict__ WhhP_b,
                                              float* __restrict__ gsc_f,
                                              float* __restrict__ gsc_b,
                                              float* __restrict__ hs_f,
                                              float* __restrict__ hs_b) {
  const int dir = blockIdx.x;
  const float* Xg   = dir ? Xg_b   : Xg_f;
  const v16bf* Bv   = (const v16bf*)(dir ? WhhP_b : WhhP_f);
  float*       gsc  = dir ? gsc_b  : gsc_f;
  float*       hs   = dir ? hs_b   : hs_f;

  __shared__ v16bf sh_hA[(NB / 16) * (HENC / 32) * 32];  // 32 KB packed-A h
  bf16* hraw = (bf16*)sh_hA;
  for (int i = threadIdx.x; i < NB * HENC; i += 512) hraw[i] = (bf16)0.0f;
  float c_reg[32];
#pragma unroll
  for (int j = 0; j < 32; ++j) c_reg[j] = 0.0f;
  __syncthreads();

  const int wave = threadIdx.x >> 5, L = threadIdx.x & 31;

  for (int t = 0; t < TENC; ++t) {
    // gates = h @ WhhT : 64 N-pair jobs, 16 waves -> 4 jobs each (2Mx2N)
    for (int q = 0; q < 4; ++q) {
      int ntb = (wave * 4 + q) << 1;
      v8f c[2][2] = {};
#pragma unroll 2
      for (int kt = 0; kt < HENC / 32; ++kt) {
        v16bf b0 = Bv[((kt * (G4 / 16) + ntb) << 5) + L];
        v16bf b1 = Bv[((kt * (G4 / 16) + ntb + 1) << 5) + L];
        v16bf a0 = sh_hA[((0 * (HENC / 32) + kt) << 5) + L];
        v16bf a1 = sh_hA[((1 * (HENC / 32) + kt) << 5) + L];
        c[0][0] = WMMA_BF16(a0, b0, c[0][0]);
        c[0][1] = WMMA_BF16(a0, b1, c[0][1]);
        c[1][0] = WMMA_BF16(a1, b0, c[1][0]);
        c[1][1] = WMMA_BF16(a1, b1, c[1][1]);
      }
#pragma unroll
      for (int im = 0; im < 2; ++im)
#pragma unroll
        for (int jn = 0; jn < 2; ++jn) {
          int n = ((ntb + jn) << 4) + (L & 15);
#pragma unroll
          for (int r = 0; r < 8; ++r) {
            int m = (im << 4) + r + ((L & 16) ? 8 : 0);
            gsc[m * G4 + n] = c[im][jn][r] + Xg[((size_t)t * NB + m) * G4 + n];
          }
        }
    }
    __threadfence_block();
    __syncthreads();
    // cell update: 16384 elems / 512 thr
#pragma unroll
    for (int j = 0; j < 32; ++j) {
      int idx = threadIdx.x + 512 * j;
      int b = idx >> 9, h = idx & (HENC - 1);
      float gi = gsc[b * G4 + h];
      float gf = gsc[b * G4 + HENC + h];
      float gg = gsc[b * G4 + 2 * HENC + h];
      float go = gsc[b * G4 + 3 * HENC + h];
      float cc = sigf(gf) * c_reg[j] + sigf(gi) * tanhf(gg);
      c_reg[j] = cc;
      float hh = sigf(go) * tanhf(cc);
      hraw[a_slot(b, h, HENC / 32)] = (bf16)hh;
      hs[((size_t)t * NB + b) * HENC + h] = hh;
    }
    __threadfence_block();
    __syncthreads();
  }
}

// ------------------------------------------------------------------
// enc_out[b,t,:] = [h_f ; h_b un-reversed], zeroed past length
// ------------------------------------------------------------------
__global__ __launch_bounds__(256) void k_combine(const int* __restrict__ lens,
                                                 const float* __restrict__ hs_f,
                                                 const float* __restrict__ hs_b,
                                                 float* __restrict__ enc_out) {
  int i = blockIdx.x * 256 + threadIdx.x;              // NB*TENC*HENC
  int h = i & (HENC - 1);
  int t = (i >> 9) & (TENC - 1);
  int b = i >> 16;
  int len = lens[b];
  bool valid = t < len;
  int rt = valid ? (len - 1 - t) : t;
  float vf = valid ? hs_f[((size_t)t  * NB + b) * HENC + h] : 0.0f;
  float vb = valid ? hs_b[((size_t)rt * NB + b) * HENC + h] : 0.0f;
  size_t base = ((size_t)b * TENC + t) * CTXD;
  enc_out[base + h]        = vf;
  enc_out[base + HENC + h] = vb;
}

// ------------------------------------------------------------------
// enc_part[b,t,a] = enc_out[b,t,:] @ attn_W[:1024, a]
// ------------------------------------------------------------------
__global__ __launch_bounds__(256) void k_encpart(const float* __restrict__ enc_out,
                                                 const float* __restrict__ attn_W,
                                                 float* __restrict__ enc_part) {
  int i = blockIdx.x * 256 + threadIdx.x;              // NB*TENC*ATT
  if (i >= NB * TENC * ATT) return;
  int a8 = i & (ATT - 1);
  int bt = i >> 3;
  float acc = 0.0f;
  for (int k = 0; k < CTXD; ++k)
    acc += enc_out[(size_t)bt * CTXD + k] * attn_W[k * ATT + a8];
  enc_part[i] = acc;
}

// ------------------------------------------------------------------
// Persistent decoder (attention + GRU), single workgroup.
// h: packed-A bf16 in LDS + f32 in registers; xin built in packed-A LDS.
// ------------------------------------------------------------------
__global__ __launch_bounds__(512) void k_decoder(const int* __restrict__ dec_toks,
                                                 const int* __restrict__ enc_lens,
                                                 const float* __restrict__ emb_tgt,
                                                 const float* __restrict__ enc_out,
                                                 const float* __restrict__ enc_part,
                                                 const float* __restrict__ attn_W,
                                                 const float* __restrict__ attn_b,
                                                 const float* __restrict__ attn_v,
                                                 const bf16* __restrict__ gWihP,
                                                 const bf16* __restrict__ gWhhP,
                                                 const float* __restrict__ gbih,
                                                 const float* __restrict__ gbhh,
                                                 float* __restrict__ gi_s,
                                                 float* __restrict__ gh_s,
                                                 float* __restrict__ H_all) {
  __shared__ v16bf sh_hA[(NB / 16) * (HDEC / 32) * 32];  // 32 KB packed h
  __shared__ v16bf sh_xinA[(NB / 16) * (KIN / 32) * 32]; // 80 KB packed xin
  __shared__ float sh_a[NB * TENC];                      // 16 KB attn weights
  __shared__ float sh_hp[NB * ATT];                      // 1 KB
  bf16* hraw = (bf16*)sh_hA;
  bf16* xraw = (bf16*)sh_xinA;
  const v16bf* BWih = (const v16bf*)gWihP;
  const v16bf* BWhh = (const v16bf*)gWhhP;

  for (int i = threadIdx.x; i < NB * HDEC; i += 512) hraw[i] = (bf16)0.0f;
  float h_reg[32];
#pragma unroll
  for (int j = 0; j < 32; ++j) h_reg[j] = 0.0f;
  __syncthreads();

  const int wave = threadIdx.x >> 5, L = threadIdx.x & 31;

  for (int td = 0; td < TDEC; ++td) {
    // h_part[b,a] = h @ attn_W[1024:, a]   (reads bf16 packed h)
    if (threadIdx.x < 256) {
      int b = threadIdx.x >> 3, a8 = threadIdx.x & 7;
      float acc = 0.0f;
      for (int k = 0; k < HDEC; ++k)
        acc += (float)hraw[a_slot(b, k, HDEC / 32)] * attn_W[(CTXD + k) * ATT + a8];
      sh_hp[b * ATT + a8] = acc;
    }
    __syncthreads();
    // scores + masked softmax: wave owns rows b = wave, wave+16
    for (int bb = 0; bb < 2; ++bb) {
      int b = wave + 16 * bb;
      int len = enc_lens[b];
      float sc[4];
#pragma unroll
      for (int jj = 0; jj < 4; ++jj) {
        int t = L + 32 * jj;
        float e = 0.0f;
#pragma unroll
        for (int a8 = 0; a8 < ATT; ++a8)
          e += attn_v[a8] * tanhf(enc_part[((size_t)b * TENC + t) * ATT + a8] +
                                  sh_hp[b * ATT + a8] + attn_b[a8]);
        sc[jj] = (t < len) ? e : -1e9f;
      }
      float mx = fmaxf(fmaxf(sc[0], sc[1]), fmaxf(sc[2], sc[3]));
      for (int o = 16; o; o >>= 1) mx = fmaxf(mx, __shfl_xor(mx, o, 32));
      float sum = 0.0f;
#pragma unroll
      for (int jj = 0; jj < 4; ++jj) { sc[jj] = expf(sc[jj] - mx); sum += sc[jj]; }
      for (int o = 16; o; o >>= 1) sum += __shfl_xor(sum, o, 32);
      float inv = 1.0f / sum;
#pragma unroll
      for (int jj = 0; jj < 4; ++jj) sh_a[b * TENC + L + 32 * jj] = sc[jj] * inv;
    }
    __syncthreads();
    // ctx + target embedding -> packed xin
    {
      int b = threadIdx.x >> 4, dl = threadIdx.x & 15;
      for (int j = 0; j < 64; ++j) {
        int d = dl + 16 * j;
        float acc = 0.0f;
        for (int t = 0; t < TENC; ++t)
          acc += sh_a[b * TENC + t] * enc_out[((size_t)b * TENC + t) * CTXD + d];
        xraw[a_slot(b, ETGT + d, KIN / 32)] = (bf16)acc;
      }
      for (int j = 0; j < 16; ++j) {
        int idx = threadIdx.x + 512 * j;               // NB*ETGT
        int bb2 = idx >> 8, e = idx & (ETGT - 1);
        int tok = dec_toks[bb2 * TDEC + td];
        xraw[a_slot(bb2, e, KIN / 32)] = (bf16)emb_tgt[(size_t)tok * ETGT + e];
      }
    }
    __syncthreads();
    // gi = xin @ gWihT + gbih : 48 N-pair jobs, 16 waves -> 3 each (2Mx2N)
    for (int q = 0; q < 3; ++q) {
      int ntb = (wave * 3 + q) << 1;
      v8f c[2][2] = {};
#pragma unroll 2
      for (int kt = 0; kt < KIN / 32; ++kt) {
        v16bf b0 = BWih[((kt * (G3 / 16) + ntb) << 5) + L];
        v16bf b1 = BWih[((kt * (G3 / 16) + ntb + 1) << 5) + L];
        v16bf a0 = sh_xinA[((0 * (KIN / 32) + kt) << 5) + L];
        v16bf a1 = sh_xinA[((1 * (KIN / 32) + kt) << 5) + L];
        c[0][0] = WMMA_BF16(a0, b0, c[0][0]);
        c[0][1] = WMMA_BF16(a0, b1, c[0][1]);
        c[1][0] = WMMA_BF16(a1, b0, c[1][0]);
        c[1][1] = WMMA_BF16(a1, b1, c[1][1]);
      }
#pragma unroll
      for (int im = 0; im < 2; ++im)
#pragma unroll
        for (int jn = 0; jn < 2; ++jn) {
          int n = ((ntb + jn) << 4) + (L & 15);
          float bias = gbih[n];
#pragma unroll
          for (int r = 0; r < 8; ++r) {
            int m = (im << 4) + r + ((L & 16) ? 8 : 0);
            gi_s[m * G3 + n] = c[im][jn][r] + bias;
          }
        }
    }
    // gh = h @ gWhhT + gbhh
    for (int q = 0; q < 3; ++q) {
      int ntb = (wave * 3 + q) << 1;
      v8f c[2][2] = {};
#pragma unroll 2
      for (int kt = 0; kt < HDEC / 32; ++kt) {
        v16bf b0 = BWhh[((kt * (G3 / 16) + ntb) << 5) + L];
        v16bf b1 = BWhh[((kt * (G3 / 16) + ntb + 1) << 5) + L];
        v16bf a0 = sh_hA[((0 * (HDEC / 32) + kt) << 5) + L];
        v16bf a1 = sh_hA[((1 * (HDEC / 32) + kt) << 5) + L];
        c[0][0] = WMMA_BF16(a0, b0, c[0][0]);
        c[0][1] = WMMA_BF16(a0, b1, c[0][1]);
        c[1][0] = WMMA_BF16(a1, b0, c[1][0]);
        c[1][1] = WMMA_BF16(a1, b1, c[1][1]);
      }
#pragma unroll
      for (int im = 0; im < 2; ++im)
#pragma unroll
        for (int jn = 0; jn < 2; ++jn) {
          int n = ((ntb + jn) << 4) + (L & 15);
          float bias = gbhh[n];
#pragma unroll
          for (int r = 0; r < 8; ++r) {
            int m = (im << 4) + r + ((L & 16) ? 8 : 0);
            gh_s[m * G3 + n] = c[im][jn][r] + bias;
          }
        }
    }
    __threadfence_block();
    __syncthreads();
    // GRU cell update: 16384 elems / 512 thr
#pragma unroll
    for (int j = 0; j < 32; ++j) {
      int idx = threadIdx.x + 512 * j;
      int b = idx >> 9, h = idx & (HDEC - 1);
      float ir = gi_s[b * G3 + h],            hr = gh_s[b * G3 + h];
      float iz = gi_s[b * G3 + HDEC + h],     hz = gh_s[b * G3 + HDEC + h];
      float in = gi_s[b * G3 + 2 * HDEC + h], hn = gh_s[b * G3 + 2 * HDEC + h];
      float r = sigf(ir + hr);
      float z = sigf(iz + hz);
      float nn = tanhf(in + r * hn);
      float hnew = (1.0f - z) * nn + z * h_reg[j];
      h_reg[j] = hnew;
      hraw[a_slot(b, h, HDEC / 32)] = (bf16)hnew;
      H_all[((size_t)td * NB + b) * HDEC + h] = hnew;
    }
    __threadfence_block();
    __syncthreads();
  }
}

// ------------------------------------------------------------------
// logits = relu(H_all @ clf_WT + clf_b) ; 4Mx2N register-blocked WMMA
// jobs: 16 Mgroups x 1000 Ngroups = 16000 waves (2000 blocks); out [b,td,v]
// ------------------------------------------------------------------
__global__ __launch_bounds__(256) void k_clf(const bf16* __restrict__ HallA,
                                             const bf16* __restrict__ WP,
                                             const float* __restrict__ bias,
                                             float* __restrict__ out) {
  const int wave = threadIdx.x >> 5, L = threadIdx.x & 31;
  const v16bf* Av = (const v16bf*)HallA;
  const v16bf* Bv = (const v16bf*)WP;
  int job = blockIdx.x * 8 + wave;
  int mg = job & 15, ng = job >> 4;
  int mtb = mg << 2, ntb = ng << 1;
  v8f c[4][2] = {};
#pragma unroll 2
  for (int kt = 0; kt < HDEC / 32; ++kt) {
    v16bf b0 = Bv[((kt * (VTGT / 16) + ntb) << 5) + L];
    v16bf b1 = Bv[((kt * (VTGT / 16) + ntb + 1) << 5) + L];
#pragma unroll
    for (int im = 0; im < 4; ++im) {
      v16bf a = Av[(((mtb + im) * (HDEC / 32) + kt) << 5) + L];
      c[im][0] = WMMA_BF16(a, b0, c[im][0]);
      c[im][1] = WMMA_BF16(a, b1, c[im][1]);
    }
  }
#pragma unroll
  for (int jn = 0; jn < 2; ++jn) {
    int n = ((ntb + jn) << 4) + (L & 15);
    float bv = bias[n];
#pragma unroll
    for (int im = 0; im < 4; ++im)
#pragma unroll
      for (int r = 0; r < 8; ++r) {
        int m = ((mtb + im) << 4) + r + ((L & 16) ? 8 : 0);  // m = td*NB + b
        int b = m & (NB - 1), td = m >> 5;
        float v = c[im][jn][r] + bv;
        out[((size_t)b * TDEC + td) * VTGT + n] = v > 0.0f ? v : 0.0f;
      }
  }
}

// ------------------------------------------------------------------
extern "C" void kernel_launch(void* const* d_in, const int* in_sizes, int n_in,
                              void* d_out, int out_size, void* d_ws, size_t ws_size,
                              hipStream_t stream) {
  (void)in_sizes; (void)n_in; (void)out_size; (void)ws_size;
  const int*   enc_in  = (const int*)d_in[0];
  const int*   enc_len = (const int*)d_in[1];
  const int*   dec_in  = (const int*)d_in[2];
  const float* emb_src = (const float*)d_in[4];
  const float* emb_tgt = (const float*)d_in[5];
  const float* Wih_f = (const float*)d_in[6];
  const float* Whh_f = (const float*)d_in[7];
  const float* bih_f = (const float*)d_in[8];
  const float* bhh_f = (const float*)d_in[9];
  const float* Wih_b = (const float*)d_in[10];
  const float* Whh_b = (const float*)d_in[11];
  const float* bih_b = (const float*)d_in[12];
  const float* bhh_b = (const float*)d_in[13];
  const float* gWih  = (const float*)d_in[14];
  const float* gWhh  = (const float*)d_in[15];
  const float* gbih  = (const float*)d_in[16];
  const float* gbhh  = (const float*)d_in[17];
  const float* attn_W = (const float*)d_in[18];
  const float* attn_b = (const float*)d_in[19];
  const float* attn_v = (const float*)d_in[20];
  const float* clf_W  = (const float*)d_in[21];
  const float* clf_b  = (const float*)d_in[22];

  char* ws = (char*)d_ws;
  size_t off = 0;
  auto wsa = [&](size_t bytes) -> void* {
    void* p = (void*)(ws + off);
    off += (bytes + 255) & ~(size_t)255;
    return p;
  };
  bf16* xA_f    = (bf16*)wsa(sizeof(bf16) * TENC * NB * ESRC);
  bf16* xA_b    = (bf16*)wsa(sizeof(bf16) * TENC * NB * ESRC);
  float* Xg_f   = (float*)wsa(sizeof(float) * TENC * NB * G4);
  float* Xg_b   = (float*)wsa(sizeof(float) * TENC * NB * G4);
  float* hs_f   = (float*)wsa(sizeof(float) * TENC * NB * HENC);
  float* hs_b   = (float*)wsa(sizeof(float) * TENC * NB * HENC);
  float* enc_out= (float*)wsa(sizeof(float) * NB * TENC * CTXD);
  float* enc_prt= (float*)wsa(sizeof(float) * NB * TENC * ATT);
  float* H_all  = (float*)wsa(sizeof(float) * TDEC * NB * HDEC);
  bf16* H_A     = (bf16*)wsa(sizeof(bf16) * TDEC * NB * HDEC);
  float* gsc_f  = (float*)wsa(sizeof(float) * NB * G4);
  float* gsc_b  = (float*)wsa(sizeof(float) * NB * G4);
  float* gi_s   = (float*)wsa(sizeof(float) * NB * G3);
  float* gh_s   = (float*)wsa(sizeof(float) * NB * G3);
  bf16* WihP_f  = (bf16*)wsa(sizeof(bf16) * ESRC * G4);
  bf16* WhhP_f  = (bf16*)wsa(sizeof(bf16) * HENC * G4);
  bf16* WihP_b  = (bf16*)wsa(sizeof(bf16) * ESRC * G4);
  bf16* WhhP_b  = (bf16*)wsa(sizeof(bf16) * HENC * G4);
  bf16* gWihP   = (bf16*)wsa(sizeof(bf16) * KIN * G3);
  bf16* gWhhP   = (bf16*)wsa(sizeof(bf16) * HDEC * G3);
  bf16* clfWP   = (bf16*)wsa(sizeof(bf16) * HDEC * VTGT);

  auto pb = [&](const float* s, bf16* d, int N, int K) {
    k_packB<<<(N * K + 255) / 256, 256, 0, stream>>>(s, d, N, K);
  };
  pb(Wih_f, WihP_f, G4, ESRC);
  pb(Whh_f, WhhP_f, G4, HENC);
  pb(Wih_b, WihP_b, G4, ESRC);
  pb(Whh_b, WhhP_b, G4, HENC);
  pb(gWih,  gWihP,  G3, KIN);
  pb(gWhh,  gWhhP,  G3, HDEC);
  pb(clf_W, clfWP,  VTGT, HDEC);

  k_embed_pack<<<(TENC * NB * ESRC) / 256, 256, 0, stream>>>(enc_in, enc_len, emb_src,
                                                             xA_f, xA_b);

  k_gemm_xg<<<512, 256, 0, stream>>>(xA_f, WihP_f, bih_f, bhh_f, Xg_f);
  k_gemm_xg<<<512, 256, 0, stream>>>(xA_b, WihP_b, bih_b, bhh_b, Xg_b);

  k_lstm<<<2, 512, 0, stream>>>(Xg_f, Xg_b, WhhP_f, WhhP_b, gsc_f, gsc_b, hs_f, hs_b);

  k_combine<<<(NB * TENC * HENC) / 256, 256, 0, stream>>>(enc_len, hs_f, hs_b, enc_out);
  k_encpart<<<(NB * TENC * ATT) / 256, 256, 0, stream>>>(enc_out, attn_W, enc_prt);

  k_decoder<<<1, 512, 0, stream>>>(dec_in, enc_len, emb_tgt, enc_out, enc_prt,
                                   attn_W, attn_b, attn_v, gWihP, gWhhP, gbih, gbhh,
                                   gi_s, gh_s, H_all);

  k_packA<<<(TDEC * NB * HDEC + 255) / 256, 256, 0, stream>>>(H_all, H_A,
                                                              TDEC * NB, HDEC);

  k_clf<<<2000, 256, 0, stream>>>(H_A, clfWP, clf_b, (float*)d_out);
}
